// RAGCN_ConvE_45775761440777
// MI455X (gfx1250) — matrix operand order repaired and checked by reading, not compiled, not run on hardware
//
#include <hip/hip_runtime.h>
#include <hip/hip_bf16.h>

// ---------------- problem constants (match reference) ----------------
constexpr int NUM_ENT  = 100000;
constexpr int NREL2    = 1000;      // 2*NUM_REL rows in relation table
constexpr int NREL2P   = 1008;      // padded to 16-row tiles
constexpr int E_TOT    = 500000;
constexpr int E_HALF   = E_TOT / 2;
constexpr int D        = 200;       // embed dim
constexpr int DP       = 224;       // D padded to multiple of 32 for WMMA K
constexpr int DT       = 208;       // D padded to 16-row tiles (B rows)
constexpr int BATCH    = 256;
constexpr int NF       = 200;
constexpr int KS       = 7;
constexpr int IH       = 20, IW = 20;     // 2*KW x KH conv input
constexpr int OH       = 14, OW = 14;
constexpr int FLATN    = NF * OH * OW;    // 39200 = 1225*32 (WMMA-exact K)
constexpr float RS     = 0.9999950000374997f; // 1/sqrt(1+1e-5)

typedef __attribute__((ext_vector_type(16))) __bf16 v16bf;
typedef __attribute__((ext_vector_type(8)))  float  v8f;

union Frag { unsigned int u[8]; uint4 q[2]; v16bf v; };

__device__ __forceinline__ unsigned short f2bf(float f) {
  unsigned int u = __float_as_uint(f);
  unsigned int r = u + 0x7FFFu + ((u >> 16) & 1u);
  return (unsigned short)(r >> 16);
}

// ---------------- generic multi-pair WMMA GEMM with fused epilogues ----------------
// C[M,N] = sum_p A_p[Mp,Kp] @ B_p^T   where B_p stored transposed [Np rows, Kp]
// Buffers are padded to whole 16-row tiles -> all fragment loads branchless.
// A frag (ISA 16x32 bf16): lane m=lane%16, hi=lane/16;
//   u[0..3] <- 16B @ row + k0 + 8*hi ; u[4..7] <- 16B @ row + k0 + 16 + 8*hi
// B frag: lane n=lane%16, hi=lane/16; u[0..3] <- 16B @ row + k0 + 8*hi ; u[4..7] <- 16B @ row + k0 + 16 + 8*hi
// EPI 0: identity      EPI 1: tanh((acc - e0[n])/3 + e1[n])
// EPI 2: relu((acc + e0[n]) * (e1[n]*RS) + e2[n])
// EPI 3: sigmoid(acc + e0[n])
template <int NP, int EPI>
__global__ __launch_bounds__(128) void gemm_bf16_wmma(
    const unsigned short* A0, const unsigned short* A1, const unsigned short* A2,
    const unsigned short* B0, const unsigned short* B1, const unsigned short* B2,
    int M, int N, int Kp, float* out,
    const float* e0, const float* e1, const float* e2) {
  int lane = threadIdx.x & 31;
  int wave = threadIdx.x >> 5;
  int m0 = blockIdx.x * 16;
  int n0 = (blockIdx.y * 4 + wave) * 16;
  if (n0 >= N) return;                       // wave-uniform
  int lo = lane & 15, hi = lane >> 4;

  size_t aoff = (size_t)(m0 + lo) * Kp + 8 * hi;
  size_t boff = (size_t)(n0 + lo) * Kp + 8 * hi;
  const unsigned short* pa0 = A0 + aoff;
  const unsigned short* pb0 = B0 + boff;
  const unsigned short* pa1 = (NP >= 2) ? A1 + aoff : nullptr;
  const unsigned short* pb1 = (NP >= 2) ? B1 + boff : nullptr;
  const unsigned short* pa2 = (NP >= 3) ? A2 + aoff : nullptr;
  const unsigned short* pb2 = (NP >= 3) ? B2 + boff : nullptr;

  v8f acc = {0.f, 0.f, 0.f, 0.f, 0.f, 0.f, 0.f, 0.f};
  for (int k = 0; k < Kp; k += 32) {
    {
      Frag a, b;
      a.q[0] = *(const uint4*)(pa0);
      a.q[1] = *(const uint4*)(pa0 + 16);
      b.q[0] = *(const uint4*)(pb0);
      b.q[1] = *(const uint4*)(pb0 + 16);
      acc = __builtin_amdgcn_wmma_f32_16x16x32_bf16(false, a.v, false, b.v,
                                                    (short)0, acc, false, false);
      pa0 += 32; pb0 += 32;
    }
    if constexpr (NP >= 2) {
      Frag a, b;
      a.q[0] = *(const uint4*)(pa1);
      a.q[1] = *(const uint4*)(pa1 + 16);
      b.q[0] = *(const uint4*)(pb1);
      b.q[1] = *(const uint4*)(pb1 + 16);
      acc = __builtin_amdgcn_wmma_f32_16x16x32_bf16(false, a.v, false, b.v,
                                                    (short)0, acc, false, false);
      pa1 += 32; pb1 += 32;
    }
    if constexpr (NP >= 3) {
      Frag a, b;
      a.q[0] = *(const uint4*)(pa2);
      a.q[1] = *(const uint4*)(pa2 + 16);
      b.q[0] = *(const uint4*)(pb2);
      b.q[1] = *(const uint4*)(pb2 + 16);
      acc = __builtin_amdgcn_wmma_f32_16x16x32_bf16(false, a.v, false, b.v,
                                                    (short)0, acc, false, false);
      pa2 += 32; pb2 += 32;
    }
  }

  // D layout: lane n=lane%16; rows m = m0 + 8*hi + v
  int n = n0 + lo;
  int mb = m0 + hi * 8;
  if (n < N) {
#pragma unroll
    for (int v = 0; v < 8; ++v) {
      int m = mb + v;
      if (m < M) {
        float x = acc[v];
        float y;
        if constexpr (EPI == 0) {
          y = x;
        } else if constexpr (EPI == 1) {
          y = tanhf((x - e0[n]) * (1.0f / 3.0f) + e1[n]);
        } else if constexpr (EPI == 2) {
          y = (x + e0[n]) * (e1[n] * RS) + e2[n];
          y = y > 0.f ? y : 0.f;
        } else {
          y = 1.0f / (1.0f + __expf(-(x + e0[n])));
        }
        out[(size_t)m * N + n] = y;
      }
    }
  }
}

// ---------------- helper kernels ----------------

// float [R,C] row-major -> bf16 [Rp,Cp] zero-padded (A-style layout)
__global__ void convpad_kernel(const float* src, unsigned short* dst,
                               int R, int C, int Rp, int Cp) {
  int t = blockIdx.x * blockDim.x + threadIdx.x;
  if (t >= Rp * Cp) return;
  int r = t / Cp, c = t - r * Cp;
  float v = (r < R && c < C) ? src[(size_t)r * C + c] : 0.f;
  dst[t] = f2bf(v);
}

// float [R,C] -> transposed bf16 [Cp rows, Rp cols]: dst[c*Rp + r] = src[r*C + c]
__global__ void convpadT_kernel(const float* src, unsigned short* dst,
                                int R, int C, int Rp, int Cp) {
  int t = blockIdx.x * blockDim.x + threadIdx.x;
  if (t >= Cp * Rp) return;
  int c = t / Rp, r = t - c * Rp;
  float v = (r < R && c < C) ? src[(size_t)r * C + c] : 0.f;
  dst[t] = f2bf(v);
}

// degree counts for both halves (dst row of edge_index)
__global__ void deg_kernel(const int* eidx, float* deg_in, float* deg_out) {
  int e = blockIdx.x * blockDim.x + threadIdx.x;
  if (e >= E_TOT) return;
  int dd = eidx[E_TOT + e];
  if (e < E_HALF) atomicAdd(&deg_in[dd], 1.0f);
  else            atomicAdd(&deg_out[dd], 1.0f);
}

// scatter-add (x[src] - r[et]) rows into sum[dst]; float4 granularity (D=200=50*4)
__global__ void edge_scatter_kernel(const int* eidx, const int* etype,
                                    const float* x, const float* r,
                                    float* sum, int off, int count) {
  int t = blockIdx.x * blockDim.x + threadIdx.x;
  if (t >= count * (D / 4)) return;
  int e = t / (D / 4);
  int q = t - e * (D / 4);
  int d4 = q * 4;
  int s  = eidx[off + e];
  int dd = eidx[E_TOT + off + e];
  int rt = etype[off + e];
  float4 xv = *(const float4*)(x + (size_t)s * D + d4);
  float4 rv = *(const float4*)(r + (size_t)rt * D + d4);
  float* o = sum + (size_t)dd * D + d4;
  atomicAdd(o + 0, xv.x - rv.x);
  atomicAdd(o + 1, xv.y - rv.y);
  atomicAdd(o + 2, xv.z - rv.z);
  atomicAdd(o + 3, xv.w - rv.w);
}

// sum/deg -> bf16 padded A matrix [NUM_ENT, DP]
__global__ void normalize_kernel(const float* sum, const float* deg,
                                 unsigned short* abf) {
  int t = blockIdx.x * blockDim.x + threadIdx.x;
  if (t >= NUM_ENT * DP) return;
  int n = t / DP, c = t - n * DP;
  float v = 0.f;
  if (c < D) {
    float dg = deg[n];
    if (dg < 1.f) dg = 1.f;
    v = sum[(size_t)n * D + c] / dg;
  }
  abf[t] = f2bf(v);
}

// loopcorr[j] = (loop_rel @ w_loop)[j]
__global__ void vecmat_kernel(const float* lr, const float* w, float* out) {
  int j = blockIdx.x * blockDim.x + threadIdx.x;
  if (j >= D) return;
  float acc = 0.f;
  for (int i = 0; i < D; ++i) acc += lr[i] * w[(size_t)i * D + j];
  out[j] = acc;
}

// gather sub/rel embeddings
__global__ void gather_kernel(const int* sub, const int* rel,
                              const float* x2, const float* r2,
                              float* sub_emb, float* rel_emb) {
  int t = blockIdx.x * blockDim.x + threadIdx.x;
  if (t >= BATCH * D) return;
  int b = t / D, d = t - b * D;
  sub_emb[t] = x2[(size_t)sub[b] * D + d];
  rel_emb[t] = r2[(size_t)rel[b] * D + d];
}

// ConvE: BN0 -> 7x7 conv -> +bias -> BN1 -> relu -> bf16 [BATCH, FLATN]
__global__ void conv_bn_relu_kernel(const float* sub_emb, const float* rel_emb,
                                    const float* conv_w, const float* conv_b,
                                    const float* bn0_g, const float* bn0_b,
                                    const float* bn1_g, const float* bn1_b,
                                    unsigned short* hconv_bf) {
  int t = blockIdx.x * blockDim.x + threadIdx.x;
  if (t >= BATCH * NF * OH * OW) return;
  int b   = t / (NF * OH * OW);
  int rem = t - b * (NF * OH * OW);
  int f   = rem / (OH * OW);
  int p   = rem - f * (OH * OW);
  int y   = p / OW;
  int xo  = p - y * OW;
  float s0 = bn0_g[0] * RS, b0 = bn0_b[0];
  const float* se = sub_emb + (size_t)b * D;
  const float* re = rel_emb + (size_t)b * D;
  const float* w  = conv_w + (size_t)f * KS * KS;
  float acc = 0.f;
#pragma unroll
  for (int ky = 0; ky < KS; ++ky) {
    int h = y + ky;
#pragma unroll
    for (int kx = 0; kx < KS; ++kx) {
      int idx = h * IW + (xo + kx);   // stk[b, h, w]: d = idx/2, c = idx%2
      float in = (idx & 1) ? re[idx >> 1] : se[idx >> 1];
      acc += (in * s0 + b0) * w[ky * KS + kx];
    }
  }
  float v = acc + conv_b[f];
  v = v * (bn1_g[f] * RS) + bn1_b[f];
  v = v > 0.f ? v : 0.f;
  hconv_bf[(size_t)b * FLATN + f * (OH * OW) + p] = f2bf(v);
}

// ---------------- host launcher ----------------
extern "C" void kernel_launch(void* const* d_in, const int* in_sizes, int n_in,
                              void* d_out, int out_size, void* d_ws, size_t ws_size,
                              hipStream_t stream) {
  const int*   sub        = (const int*)d_in[0];
  const int*   rel        = (const int*)d_in[1];
  const int*   eidx       = (const int*)d_in[2];
  const int*   etype      = (const int*)d_in[3];
  const float* init_embed = (const float*)d_in[4];
  const float* init_rel   = (const float*)d_in[5];
  const float* loop_rel1  = (const float*)d_in[6];
  const float* loop_rel2  = (const float*)d_in[7];
  const float* w_in1  = (const float*)d_in[8];
  const float* w_out1 = (const float*)d_in[9];
  const float* w_loop1= (const float*)d_in[10];
  const float* w_rel1 = (const float*)d_in[11];
  const float* b1     = (const float*)d_in[12];
  const float* w_in2  = (const float*)d_in[13];
  const float* w_out2 = (const float*)d_in[14];
  const float* w_loop2= (const float*)d_in[15];
  const float* w_rel2 = (const float*)d_in[16];
  const float* b2     = (const float*)d_in[17];
  const float* conv_w = (const float*)d_in[18];
  const float* conv_b = (const float*)d_in[19];
  const float* bn0_g  = (const float*)d_in[20];
  const float* bn0_b  = (const float*)d_in[21];
  const float* bn1_g  = (const float*)d_in[22];
  const float* bn1_b  = (const float*)d_in[23];
  const float* bn2_g  = (const float*)d_in[24];
  const float* bn2_b  = (const float*)d_in[25];
  const float* fc_w   = (const float*)d_in[26];
  const float* fc_b   = (const float*)d_in[27];
  const float* ent_bias = (const float*)d_in[28];
  float* out = (float*)d_out;

  // ---- workspace carve-up (256B aligned) ----
  char* p = (char*)d_ws;
  auto alloc = [&](size_t bytes) -> void* {
    void* r = (void*)p;
    p += (bytes + 255) & ~(size_t)255;
    return r;
  };
  float* sum      = (float*)alloc((size_t)NUM_ENT * D * 4);
  float* deg_in   = (float*)alloc((size_t)NUM_ENT * 4);
  float* deg_out  = (float*)alloc((size_t)NUM_ENT * 4);
  float* x1       = (float*)alloc((size_t)NUM_ENT * D * 4);
  float* x2       = (float*)alloc((size_t)NUM_ENT * D * 4);
  unsigned short* xbf    = (unsigned short*)alloc((size_t)NUM_ENT * DP * 2);
  unsigned short* ainbf  = (unsigned short*)alloc((size_t)NUM_ENT * DP * 2);
  unsigned short* aoutbf = (unsigned short*)alloc((size_t)NUM_ENT * DP * 2);
  unsigned short* wbt[8];                               // transposed weights [DT, DP]
  for (int i = 0; i < 8; ++i) wbt[i] = (unsigned short*)alloc((size_t)DT * DP * 2);
  unsigned short* rbf   = (unsigned short*)alloc((size_t)NREL2P * DP * 2);
  float* r1             = (float*)alloc((size_t)NREL2 * D * 4);
  float* r2             = (float*)alloc((size_t)NREL2 * D * 4);
  float* lc1            = (float*)alloc(D * 4);
  float* lc2            = (float*)alloc(D * 4);
  float* sub_emb        = (float*)alloc((size_t)BATCH * D * 4);
  float* rel_emb        = (float*)alloc((size_t)BATCH * D * 4);
  unsigned short* hconv = (unsigned short*)alloc((size_t)BATCH * FLATN * 2);
  unsigned short* fcwbt = (unsigned short*)alloc((size_t)DT * FLATN * 2); // fc_w^T
  float* hfc            = (float*)alloc((size_t)BATCH * D * 4);
  unsigned short* hbf   = (unsigned short*)alloc((size_t)BATCH * DP * 2);

  const int TB = 256;
  auto blocks = [](long long n, int tb) { return (unsigned)((n + tb - 1) / tb); };

  // ---- weights -> transposed bf16 [DT rows, DP cols] ----
  const float* wsrc[8] = {w_in1, w_out1, w_loop1, w_rel1, w_in2, w_out2, w_loop2, w_rel2};
  for (int i = 0; i < 8; ++i)
    convpadT_kernel<<<blocks((long long)DT * DP, TB), TB, 0, stream>>>(wsrc[i], wbt[i], D, D, DP, DT);
  // fc_w [FLATN, D] -> fc_w^T bf16 [DT rows, FLATN cols]
  convpadT_kernel<<<blocks((long long)DT * FLATN, TB), TB, 0, stream>>>(fc_w, fcwbt, FLATN, D, FLATN, DT);

  vecmat_kernel<<<1, 256, 0, stream>>>(loop_rel1, w_loop1, lc1);
  vecmat_kernel<<<1, 256, 0, stream>>>(loop_rel2, w_loop2, lc2);

  // ---- degrees (same for both layers) ----
  hipMemsetAsync(deg_in, 0, (size_t)NUM_ENT * 4, stream);
  hipMemsetAsync(deg_out, 0, (size_t)NUM_ENT * 4, stream);
  deg_kernel<<<blocks(E_TOT, TB), TB, 0, stream>>>(eidx, deg_in, deg_out);

  const int NT_D  = (D + 15) / 16;                 // 13 N-tiles
  dim3 gLayer(NUM_ENT / 16, (NT_D + 3) / 4);       // (6250, 4)
  dim3 gRel(NREL2P / 16, (NT_D + 3) / 4);          // (63, 4)

  // ---- two RAGCN layers ----
  for (int layer = 0; layer < 2; ++layer) {
    const float* xin = (layer == 0) ? init_embed : x1;
    const float* rin = (layer == 0) ? init_rel : r1;
    float* xo = (layer == 0) ? x1 : x2;
    float* ro = (layer == 0) ? r1 : r2;
    const unsigned short* Win  = wbt[layer * 4 + 0];
    const unsigned short* Wout = wbt[layer * 4 + 1];
    const unsigned short* Wlp  = wbt[layer * 4 + 2];
    const unsigned short* Wrl  = wbt[layer * 4 + 3];
    const float* lc = (layer == 0) ? lc1 : lc2;
    const float* bb = (layer == 0) ? b1 : b2;

    convpad_kernel<<<blocks((long long)NUM_ENT * DP, TB), TB, 0, stream>>>(xin, xbf, NUM_ENT, D, NUM_ENT, DP);
    convpad_kernel<<<blocks((long long)NREL2P * DP, TB), TB, 0, stream>>>(rin, rbf, NREL2, D, NREL2P, DP);

    // in-half aggregate
    hipMemsetAsync(sum, 0, (size_t)NUM_ENT * D * 4, stream);
    edge_scatter_kernel<<<blocks((long long)E_HALF * (D / 4), TB), TB, 0, stream>>>(eidx, etype, xin, rin, sum, 0, E_HALF);
    normalize_kernel<<<blocks((long long)NUM_ENT * DP, TB), TB, 0, stream>>>(sum, deg_in, ainbf);
    // out-half aggregate
    hipMemsetAsync(sum, 0, (size_t)NUM_ENT * D * 4, stream);
    edge_scatter_kernel<<<blocks((long long)E_HALF * (D / 4), TB), TB, 0, stream>>>(eidx, etype, xin, rin, sum, E_HALF, E_HALF);
    normalize_kernel<<<blocks((long long)NUM_ENT * DP, TB), TB, 0, stream>>>(sum, deg_out, aoutbf);

    // x_next = tanh((Ain@Win + Aout@Wout + X@Wloop - lc)/3 + b)
    gemm_bf16_wmma<3, 1><<<gLayer, 128, 0, stream>>>(
        ainbf, aoutbf, xbf, Win, Wout, Wlp, NUM_ENT, D, DP, xo, lc, bb, nullptr);
    // r_next = r @ w_rel
    gemm_bf16_wmma<1, 0><<<gRel, 128, 0, stream>>>(
        rbf, nullptr, nullptr, Wrl, nullptr, nullptr, NREL2, D, DP, ro,
        nullptr, nullptr, nullptr);
  }

  // ---- ConvE head ----
  convpad_kernel<<<blocks((long long)NUM_ENT * DP, TB), TB, 0, stream>>>(x2, xbf, NUM_ENT, D, NUM_ENT, DP); // x2 -> bf16 (logits B^T)
  gather_kernel<<<blocks((long long)BATCH * D, TB), TB, 0, stream>>>(sub, rel, x2, r2, sub_emb, rel_emb);
  conv_bn_relu_kernel<<<blocks((long long)BATCH * NF * OH * OW, TB), TB, 0, stream>>>(
      sub_emb, rel_emb, conv_w, conv_b, bn0_g, bn0_b, bn1_g, bn1_b, hconv);

  // h = relu(bn2(hconv @ fc_w + fc_b))
  dim3 gFc(BATCH / 16, (NT_D + 3) / 4);
  gemm_bf16_wmma<1, 2><<<gFc, 128, 0, stream>>>(
      hconv, nullptr, nullptr, fcwbt, nullptr, nullptr, BATCH, D, FLATN, hfc,
      fc_b, bn2_g, bn2_b);
  convpad_kernel<<<blocks((long long)BATCH * DP, TB), TB, 0, stream>>>(hfc, hbf, BATCH, D, BATCH, DP);

  // logits = sigmoid(h @ x2^T + ent_bias)   [256, 100000]
  const int NT_E = NUM_ENT / 16;                   // 6250
  dim3 gLog(BATCH / 16, (NT_E + 3) / 4);
  gemm_bf16_wmma<1, 3><<<gLog, 128, 0, stream>>>(
      hbf, nullptr, nullptr, xbf, nullptr, nullptr, BATCH, NUM_ENT, DP, out,
      ent_bias, nullptr, nullptr);
}
